// Mamba_74861279969329
// MI455X (gfx1250) — matrix operand order repaired and checked by reading, not compiled, not run on hardware
//
#include <hip/hip_runtime.h>
#include <hip/hip_bf16.h>

// ---------------------------------------------------------------------------
// Mamba forward (2 layers) for gfx1250.
//   - f16 WMMA GEMMs (128x64x64 block tile, double-buffered LDS,
//     async global->LDS via global_load_async_to_lds_b128)
//   - VALU selective scan (one thread per channel, 16-state in registers)
// ---------------------------------------------------------------------------

typedef __attribute__((ext_vector_type(16))) _Float16 v16h;
typedef __attribute__((ext_vector_type(8)))  _Float16 v8h;
typedef __attribute__((ext_vector_type(4)))  _Float16 v4h;
typedef __attribute__((ext_vector_type(8)))  float    v8f;

#define LSEQ    2048
#define DMODEL  1024
#define DINNER  2048
#define DSTATE  16
#define DTRANK  128
#define DPROJ   4096   // 2*DINNER
#define XDBLW   160    // DTRANK + 2*DSTATE

#if defined(__HIP_DEVICE_COMPILE__) && __has_builtin(__builtin_amdgcn_global_load_async_to_lds_b128)
#define HAS_ASYNC 1
#else
#define HAS_ASYNC 0
#endif

#if HAS_ASYNC
// Builtin signature (from clang diagnostics): (v4i addrspace(1)*, v4i addrspace(3)*, imm, imm)
typedef int gi4 __attribute__((vector_size(16)));
typedef __attribute__((address_space(1))) gi4 gi4_glob;
typedef __attribute__((address_space(3))) gi4 gi4_lds;
// generic->AS1: bit-identical for global memory; generic->AS3: low 32 bits
// hold the LDS byte address.
#define AS1(p) ((gi4_glob*)(uintptr_t)(p))
#define AS3(p) ((gi4_lds*)(uint32_t)(uintptr_t)(p))
#endif

__device__ __forceinline__ void wait_async0() {
#if HAS_ASYNC
    asm volatile("s_wait_asynccnt 0x0" ::: "memory");
#endif
}

// ---------------------------------------------------------------------------
// RMSNorm: one block per row, 256 threads, wave32 shuffle reduction.
// ---------------------------------------------------------------------------
__global__ __launch_bounds__(256)
void rmsnorm_kernel(const float* __restrict__ x, const float* __restrict__ w,
                    float* __restrict__ out, int D) {
    const int row = blockIdx.x;
    const float* xr = x + (size_t)row * D;
    float s = 0.f;
    for (int c = threadIdx.x; c < D; c += blockDim.x) {
        float v = xr[c];
        s += v * v;
    }
    for (int off = 16; off > 0; off >>= 1) s += __shfl_xor(s, off, 32);
    __shared__ float red[8];
    const int lane = threadIdx.x & 31;
    const int wv   = threadIdx.x >> 5;
    if (lane == 0) red[wv] = s;
    __syncthreads();
    if (wv == 0) {
        float t = (lane < 8) ? red[lane] : 0.f;
        for (int off = 4; off > 0; off >>= 1) t += __shfl_xor(t, off, 32);
        if (lane == 0) red[0] = rsqrtf(t / (float)D + 1e-5f);
    }
    __syncthreads();
    const float scale = red[0];
    float* orow = out + (size_t)row * D;
    for (int c = threadIdx.x; c < D; c += blockDim.x)
        orow[c] = xr[c] * scale * w[c];
}

// ---------------------------------------------------------------------------
// f32 -> f16 convert, vectorized float4 -> 4x f16.
// ---------------------------------------------------------------------------
__global__ __launch_bounds__(256)
void cvt_f16_kernel(const float* __restrict__ src, _Float16* __restrict__ dst,
                    int count4) {
    const int i = blockIdx.x * blockDim.x + threadIdx.x;
    if (i >= count4) return;
    const float4 v = ((const float4*)src)[i];
    v4h h = {(_Float16)v.x, (_Float16)v.y, (_Float16)v.z, (_Float16)v.w};
    ((v4h*)dst)[i] = h;
}

// ---------------------------------------------------------------------------
// Weight transpose + convert:  src (K x N, f32) -> dst (N x K, f16).
// 32x32 LDS tile, block (32,8).  K % 32 == 0, N % 32 == 0.
// ---------------------------------------------------------------------------
__global__ __launch_bounds__(256)
void transpose_cvt_kernel(const float* __restrict__ src, _Float16* __restrict__ dst,
                          int K, int N) {
    __shared__ float tile[32][33];
    const int n = blockIdx.x * 32 + threadIdx.x;
    const int k0 = blockIdx.y * 32;
    for (int j = threadIdx.y; j < 32; j += 8)
        tile[j][threadIdx.x] = src[(size_t)(k0 + j) * N + n];
    __syncthreads();
    const int kk = threadIdx.x;
    for (int j = threadIdx.y; j < 32; j += 8)
        dst[(size_t)(blockIdx.x * 32 + j) * K + k0 + kk] = (_Float16)tile[kk][j];
}

// ---------------------------------------------------------------------------
// Causal depthwise conv (K=4) + SiLU.
// ---------------------------------------------------------------------------
__global__ __launch_bounds__(256)
void conv_silu_kernel(const float* __restrict__ proj, int ldp,
                      const float* __restrict__ cw, const float* __restrict__ cb,
                      float* __restrict__ u, int L, int Di) {
    const int idx = blockIdx.x * blockDim.x + threadIdx.x;
    if (idx >= L * Di) return;
    const int t = idx / Di;
    const int d = idx - t * Di;
    float acc = cb[d];
#pragma unroll
    for (int k = 0; k < 4; ++k) {
        const int ts = t + k - 3;
        if (ts >= 0) acc += proj[(size_t)ts * ldp + d] * cw[d * 4 + k];
    }
    u[idx] = acc / (1.f + __expf(-acc));   // silu
}

// ---------------------------------------------------------------------------
// Selective scan fused with  y = (scan + D*u) * silu(z).
// One thread per channel d; B_t/C_t fetched as 4x b128 each.
// ---------------------------------------------------------------------------
__global__ __launch_bounds__(256)
void scan_kernel(const float* __restrict__ u, const float* __restrict__ delta,
                 const float* __restrict__ A_log, const float* __restrict__ xdbl,
                 const float* __restrict__ D_p, const float* __restrict__ proj,
                 float* __restrict__ y, int L, int Di, int ldx, int ldp) {
    const int d = blockIdx.x * blockDim.x + threadIdx.x;
    if (d >= Di) return;
    float a[DSTATE], h[DSTATE];
#pragma unroll
    for (int n = 0; n < DSTATE; ++n) {
        a[n] = -__expf(A_log[(size_t)d * DSTATE + n]);
        h[n] = 0.f;
    }
    const float Dp = D_p[d];
    for (int t = 0; t < L; ++t) {
        const float dt = delta[(size_t)t * Di + d];
        const float ut = u[(size_t)t * Di + d];
        const float du = dt * ut;
        const float4* bc = (const float4*)(xdbl + (size_t)t * ldx + DTRANK);
        float bn[DSTATE], cn[DSTATE];
#pragma unroll
        for (int q = 0; q < 4; ++q) {
            const float4 bv = bc[q];
            const float4 cv = bc[q + 4];
            bn[q * 4 + 0] = bv.x; bn[q * 4 + 1] = bv.y;
            bn[q * 4 + 2] = bv.z; bn[q * 4 + 3] = bv.w;
            cn[q * 4 + 0] = cv.x; cn[q * 4 + 1] = cv.y;
            cn[q * 4 + 2] = cv.z; cn[q * 4 + 3] = cv.w;
        }
        float acc = 0.f;
#pragma unroll
        for (int n = 0; n < DSTATE; ++n) {
            h[n] = __expf(dt * a[n]) * h[n] + du * bn[n];
            acc += h[n] * cn[n];
        }
        float yv = acc + Dp * ut;
        const float zv = proj[(size_t)t * ldp + Di + d];
        yv *= zv / (1.f + __expf(-zv));    // * silu(z)
        y[(size_t)t * Di + d] = yv;
    }
}

// ---------------------------------------------------------------------------
// WMMA GEMM (f16 inputs, f32 accumulate):
//   C(MxN, f32) = A(MxK, f16, row-major) * Bt(NxK, f16, K-contiguous)^T
// Block tile 128x64, K-step 64, 256 threads = 8 waves in 4x2 layout,
// each wave owns a 32x32 accumulator block -> 8 WMMAs per staged tile.
// Double-buffered LDS, async global->LDS staging.
// EPI: 0 = store, 1 = softplus(acc + bias[n]), 2 = C += acc (residual).
// Requires: M % 128 == 0, K % 64 == 0; N ragged (guarded).
// ---------------------------------------------------------------------------
__device__ __forceinline__ v16h frag16(const _Float16* p0, const _Float16* p1) {
    const v8h lo = *(const v8h*)p0;
    const v8h hi = *(const v8h*)p1;
    return __builtin_shufflevector(lo, hi,
        0, 1, 2, 3, 4, 5, 6, 7, 8, 9, 10, 11, 12, 13, 14, 15);
}

template <int EPI>
__global__ __launch_bounds__(256)
void gemm_wmma_kernel(const _Float16* __restrict__ A, int lda,
                      const _Float16* __restrict__ Bt, int ldb,
                      float* __restrict__ C, int ldc,
                      int M, int N, int K, const float* __restrict__ bias) {
    constexpr int BM = 128, BN = 64, BK = 64, LST = BK + 8;  // padded LDS row
    __shared__ __align__(16) _Float16 As[2][BM * LST];
    __shared__ __align__(16) _Float16 Bs[2][BN * LST];

    const int m0 = blockIdx.y * BM;
    const int n0 = blockIdx.x * BN;
    const int tid  = threadIdx.x;
    const int lane = tid & 31;
    const int wv   = tid >> 5;
    const int ms   = (wv & 3) * 32;        // wave row block (4 waves down)
    const int ns   = (wv >> 2) * 32;       // wave col block (2 waves across)
    const int r    = lane & 15;
    const int half = lane >> 4;

    v8f c[2][2];
#pragma unroll
    for (int i = 0; i < 2; ++i)
#pragma unroll
        for (int j = 0; j < 2; ++j)
            c[i][j] = (v8f){0.f, 0.f, 0.f, 0.f, 0.f, 0.f, 0.f, 0.f};

    auto stageA = [&](int k0, int buf) {
#pragma unroll
        for (int rep = 0; rep < 4; ++rep) {
            const int q   = tid + rep * 256;
            const int row = q >> 3;
            const int cc  = (q & 7) * 8;         // 8 f16 = 16B chunk
            const _Float16* g = A + (size_t)(m0 + row) * lda + k0 + cc;
            _Float16* l = &As[buf][row * LST + cc];
#if HAS_ASYNC
            __builtin_amdgcn_global_load_async_to_lds_b128(AS1(g), AS3(l), 0, 0);
#else
            *(v8h*)l = *(const v8h*)g;
#endif
        }
    };
    auto stageB = [&](int k0, int buf) {
#pragma unroll
        for (int rep = 0; rep < 2; ++rep) {
            const int q   = tid + rep * 256;
            const int row = q >> 3;
            const int cc  = (q & 7) * 8;
            _Float16* l = &Bs[buf][row * LST + cc];
            if (n0 + row < N) {
                const _Float16* g = Bt + (size_t)(n0 + row) * ldb + k0 + cc;
#if HAS_ASYNC
                __builtin_amdgcn_global_load_async_to_lds_b128(AS1(g), AS3(l), 0, 0);
#else
                *(v8h*)l = *(const v8h*)g;
#endif
            } else {
                *(v8h*)l = (v8h){0, 0, 0, 0, 0, 0, 0, 0};
            }
        }
    };

    stageA(0, 0);
    stageB(0, 0);
    wait_async0();
    __syncthreads();

    int p = 0;
    for (int k0 = 0; k0 < K; k0 += BK) {
        const int kn = k0 + BK;
        if (kn < K) { stageA(kn, p ^ 1); stageB(kn, p ^ 1); }

#pragma unroll
        for (int kk = 0; kk < BK; kk += 32) {
            // A fragments: elems j<8 -> K=half*8+j ; j>=8 -> K=16+half*8+(j-8)
            const _Float16* a0b = &As[p][(ms + r) * LST + kk];
            const _Float16* a1b = &As[p][(ms + 16 + r) * LST + kk];
            const v16h a0 = frag16(a0b + half * 8, a0b + 16 + half * 8);
            const v16h a1 = frag16(a1b + half * 8, a1b + 16 + half * 8);
            // B fragments: elems j<8 -> K=half*8+j ; j>=8 -> K=16+half*8+(j-8)
            const _Float16* b0b = &Bs[p][(ns + r) * LST + kk];
            const _Float16* b1b = &Bs[p][(ns + 16 + r) * LST + kk];
            const v16h b0 = frag16(b0b + half * 8, b0b + 16 + half * 8);
            const v16h b1 = frag16(b1b + half * 8, b1b + 16 + half * 8);

            c[0][0] = __builtin_amdgcn_wmma_f32_16x16x32_f16(false, a0, false, b0,
                                                             (short)0, c[0][0], false, false);
            c[0][1] = __builtin_amdgcn_wmma_f32_16x16x32_f16(false, a0, false, b1,
                                                             (short)0, c[0][1], false, false);
            c[1][0] = __builtin_amdgcn_wmma_f32_16x16x32_f16(false, a1, false, b0,
                                                             (short)0, c[1][0], false, false);
            c[1][1] = __builtin_amdgcn_wmma_f32_16x16x32_f16(false, a1, false, b1,
                                                             (short)0, c[1][1], false, false);
        }
        wait_async0();
        __syncthreads();
        p ^= 1;
    }

    // C/D layout: lane&15 -> N column, vgpr v -> row v (lanes<16) | v+8
#pragma unroll
    for (int mi = 0; mi < 2; ++mi) {
#pragma unroll
        for (int ni = 0; ni < 2; ++ni) {
            const int ncol = n0 + ns + ni * 16 + r;
            if (ncol >= N) continue;
            const v8f cc = c[mi][ni];
#pragma unroll
            for (int v = 0; v < 8; ++v) {
                const int mrow = m0 + ms + mi * 16 + half * 8 + v;
                const size_t idx = (size_t)mrow * ldc + ncol;
                const float acc = cc[v];
                if (EPI == 0) {
                    C[idx] = acc;
                } else if (EPI == 1) {
                    const float s = acc + bias[ncol];
                    C[idx] = (s > 20.f) ? s : __logf(1.f + __expf(s));
                } else {
                    C[idx] += acc;
                }
            }
        }
    }
}

// ---------------------------------------------------------------------------
// Host-side orchestration.
// ---------------------------------------------------------------------------
extern "C" void kernel_launch(void* const* d_in, const int* in_sizes, int n_in,
                              void* d_out, int out_size, void* d_ws, size_t ws_size,
                              hipStream_t stream) {
    (void)in_sizes; (void)n_in; (void)out_size; (void)ws_size;

    const float* x_in    = (const float*)d_in[0];
    const float* norm_w  = (const float*)d_in[1];
    const float* in_w    = (const float*)d_in[2];
    const float* conv_w  = (const float*)d_in[3];
    const float* conv_b  = (const float*)d_in[4];
    const float* xproj_w = (const float*)d_in[5];
    const float* dt_w    = (const float*)d_in[6];
    const float* dt_b    = (const float*)d_in[7];
    const float* A_log   = (const float*)d_in[8];
    const float* D_p     = (const float*)d_in[9];
    const float* out_w   = (const float*)d_in[10];
    const float* normf_w = (const float*)d_in[11];

    char* ws = (char*)d_ws;
    size_t off = 0;
    auto alloc = [&](size_t bytes) {
        void* p = (void*)(ws + off);
        off += (bytes + 255) & ~(size_t)255;
        return p;
    };
    float* X     = (float*)alloc((size_t)LSEQ * DMODEL * 4);   // residual stream
    float* XN    = (float*)alloc((size_t)LSEQ * DMODEL * 4);
    float* PROJ  = (float*)alloc((size_t)LSEQ * DPROJ  * 4);   // u_raw | z
    float* U     = (float*)alloc((size_t)LSEQ * DINNER * 4);
    float* DELTA = (float*)alloc((size_t)LSEQ * DINNER * 4);
    float* Y     = (float*)alloc((size_t)LSEQ * DINNER * 4);
    float* XDBL  = (float*)alloc((size_t)LSEQ * XDBLW  * 4);
    _Float16* AH = (_Float16*)alloc((size_t)LSEQ * DINNER * 2);   // f16 activations
    _Float16* WT = (_Float16*)alloc((size_t)DPROJ * DMODEL * 2);  // f16 W^T (max)

    (void)hipMemcpyAsync(X, x_in, (size_t)LSEQ * DMODEL * 4,
                         hipMemcpyDeviceToDevice, stream);

    const dim3 blk(256);
    const dim3 tblk(32, 8);
    for (int i = 0; i < 2; ++i) {
        const float* nw   = norm_w  + (size_t)i * DMODEL;
        const float* iw   = in_w    + (size_t)i * DMODEL * DPROJ;
        const float* cw   = conv_w  + (size_t)i * DINNER * 4;
        const float* cb   = conv_b  + (size_t)i * DINNER;
        const float* xpw  = xproj_w + (size_t)i * DINNER * XDBLW;
        const float* dtw  = dt_w    + (size_t)i * DTRANK * DINNER;
        const float* dtb  = dt_b    + (size_t)i * DINNER;
        const float* alog = A_log   + (size_t)i * DINNER * DSTATE;
        const float* dp   = D_p     + (size_t)i * DINNER;
        const float* ow   = out_w   + (size_t)i * DINNER * DMODEL;

        // ---- in_proj: PROJ = XN @ in_w ----
        rmsnorm_kernel<<<LSEQ, blk, 0, stream>>>(X, nw, XN, DMODEL);
        cvt_f16_kernel<<<(LSEQ * DMODEL / 4) / 256, blk, 0, stream>>>(
            XN, AH, LSEQ * DMODEL / 4);
        transpose_cvt_kernel<<<dim3(DPROJ / 32, DMODEL / 32), tblk, 0, stream>>>(
            iw, WT, DMODEL, DPROJ);
        gemm_wmma_kernel<0><<<dim3(DPROJ / 64, LSEQ / 128), blk, 0, stream>>>(
            AH, DMODEL, WT, DMODEL, PROJ, DPROJ, LSEQ, DPROJ, DMODEL, nullptr);

        // ---- u = silu(causal_dw_conv(proj[:, :Di])) ----
        conv_silu_kernel<<<(LSEQ * DINNER) / 256, blk, 0, stream>>>(
            PROJ, DPROJ, cw, cb, U, LSEQ, DINNER);

        // ---- x_dbl = U @ xproj_w (N = 160, ragged) ----
        cvt_f16_kernel<<<(LSEQ * DINNER / 4) / 256, blk, 0, stream>>>(
            U, AH, LSEQ * DINNER / 4);
        transpose_cvt_kernel<<<dim3(XDBLW / 32, DINNER / 32), tblk, 0, stream>>>(
            xpw, WT, DINNER, XDBLW);
        gemm_wmma_kernel<0><<<dim3((XDBLW + 63) / 64, LSEQ / 128), blk, 0, stream>>>(
            AH, DINNER, WT, DINNER, XDBL, XDBLW, LSEQ, XDBLW, DINNER, nullptr);

        // ---- delta = softplus(x_dbl[:, :128] @ dt_w + dt_b) ----
        cvt_f16_kernel<<<(LSEQ * XDBLW / 4) / 256, blk, 0, stream>>>(
            XDBL, AH, LSEQ * XDBLW / 4);
        transpose_cvt_kernel<<<dim3(DINNER / 32, DTRANK / 32), tblk, 0, stream>>>(
            dtw, WT, DTRANK, DINNER);
        gemm_wmma_kernel<1><<<dim3(DINNER / 64, LSEQ / 128), blk, 0, stream>>>(
            AH, XDBLW, WT, DTRANK, DELTA, DINNER, LSEQ, DINNER, DTRANK, dtb);

        // ---- selective scan fused with (+D*u) * silu(z) ----
        scan_kernel<<<DINNER / 256, blk, 0, stream>>>(
            U, DELTA, alog, XDBL, dp, PROJ, Y, LSEQ, DINNER, XDBLW, DPROJ);

        // ---- X += Y @ out_w (residual epilogue) ----
        cvt_f16_kernel<<<(LSEQ * DINNER / 4) / 256, blk, 0, stream>>>(
            Y, AH, LSEQ * DINNER / 4);
        transpose_cvt_kernel<<<dim3(DMODEL / 32, DINNER / 32), tblk, 0, stream>>>(
            ow, WT, DINNER, DMODEL);
        gemm_wmma_kernel<2><<<dim3(DMODEL / 64, LSEQ / 128), blk, 0, stream>>>(
            AH, DINNER, WT, DINNER, X, DMODEL, LSEQ, DMODEL, DINNER, nullptr);
    }

    rmsnorm_kernel<<<LSEQ, blk, 0, stream>>>(X, normf_w, (float*)d_out, DMODEL);
}
